// CrissCrossAttentionOps_76227079569747
// MI455X (gfx1250) — matrix operand order repaired and checked by reading, not compiled, not run on hardware
//
#include <hip/hip_runtime.h>

typedef __bf16 bf16_t;
typedef __attribute__((ext_vector_type(8)))  __bf16 v8bf;
typedef __attribute__((ext_vector_type(16))) __bf16 v16bf;
typedef __attribute__((ext_vector_type(8)))  float  v8f;
typedef __attribute__((ext_vector_type(4)))  unsigned v4u;
typedef __attribute__((ext_vector_type(8)))  int      v8i;
typedef __attribute__((ext_vector_type(4)))  int      v4i;

#define NINF (-3.0e38f)
#define LDP 72   // padded LDS row stride (elements): 144 B = 128 B data + 16 B pad

__device__ __forceinline__ bf16_t f2bf(float f) {
  unsigned u = __builtin_bit_cast(unsigned, f);
  unsigned r = (u + 0x7FFFu + ((u >> 16) & 1u)) >> 16;
  unsigned short s = (unsigned short)r;
  return __builtin_bit_cast(bf16_t, s);
}

__device__ __forceinline__ unsigned pk2(float a, float b) {
  unsigned short lo = __builtin_bit_cast(unsigned short, f2bf(a));
  unsigned short hi = __builtin_bit_cast(unsigned short, f2bf(b));
  return (unsigned)lo | ((unsigned)hi << 16);
}

__device__ __forceinline__ v16bf join16(v8bf lo, v8bf hi) {
  return __builtin_shufflevector(lo, hi, 0, 1, 2, 3, 4, 5, 6, 7,
                                 8, 9, 10, 11, 12, 13, 14, 15);
}

// A fragment (16x32 bf16, ISA 7.12.2): lane(m,half) takes k = 8h..8h+7 and
// 16+8h..16+8h+7 from a k-contiguous row. Two 16-byte loads.
__device__ __forceinline__ v16bf afrag(const bf16_t* row, int k0, int half) {
  v8bf lo = *(const v8bf*)(row + k0 + 8 * half);
  v8bf hi = *(const v8bf*)(row + k0 + 16 + 8 * half);
  return join16(lo, hi);
}

// B fragment (32x16 bf16): lane(n,half) takes k = 8h..8h+7 and
// 16+8h..16+8h+7 from a k-contiguous [N][K] row. Two 16-byte loads.
__device__ __forceinline__ v16bf bfrag(const bf16_t* row, int k0, int half) {
  v8bf lo = *(const v8bf*)(row + k0 + 8 * half);
  v8bf hi = *(const v8bf*)(row + k0 + 16 + 8 * half);
  return join16(lo, hi);
}

__device__ __forceinline__ v8f wmma_bf16(v16bf a, v16bf b, v8f c) {
  return __builtin_amdgcn_wmma_f32_16x16x32_bf16(false, a, false, b,
                                                 (short)0, c, false, false);
}

// ---------------------------------------------------------------------------
// TDM 2-D tile load: 64 rows x rowlen bf16, row stride stride_elems, into LDS
// with 16 B padding per 128 B row (matches [64][LDP] arrays). D# per ISA ch.8.
// Issue from a single wave (TDM ignores EXEC; guard by control flow), then
// s_wait_tensorcnt 0 before the workgroup barrier.
// ---------------------------------------------------------------------------
__device__ __forceinline__ void tdm_load_2d(const void* gptr, void* lds,
                                            unsigned rows, unsigned rowlen,
                                            unsigned stride_elems) {
  unsigned long long ga = (unsigned long long)(size_t)gptr;
  unsigned lo = (unsigned)(size_t)lds;              // low 32 bits = LDS offset
  v4u g0;
  g0.x = 1u;                                        // count=1, user descriptor
  g0.y = lo;                                        // lds_addr
  g0.z = (unsigned)(ga & 0xFFFFFFFFu);              // global_addr[31:0]
  g0.w = (unsigned)((ga >> 32) & 0x01FFFFFFu) | 0x80000000u;  // ga[56:32], type=2
  v8i g1;
  g1[0] = (int)((1u << 16)          // data_size = 2 bytes
              | (1u << 20)          // pad_enable
              | (4u << 22)          // pad_interval: 32 DWORDs (128 B)
              | (3u << 25));        // pad_amount: 4 DWORDs (16 B)
  g1[1] = (int)(rowlen << 16);      // tensor_dim0[15:0] -> d1[31:16]
  g1[2] = (int)(rows << 16);        // tensor_dim0 hi=0, tensor_dim1[15:0]
  g1[3] = (int)(rowlen << 16);      // tensor_dim1 hi=0, tile_dim0
  g1[4] = (int)rows;                // tile_dim1 (tile_dim2 = 0)
  g1[5] = (int)stride_elems;        // tensor_dim0_stride lo32 (data_size units)
  g1[6] = 0;
  g1[7] = 0;
  v4i z4 = {0, 0, 0, 0};
  v8i z8 = {0, 0, 0, 0, 0, 0, 0, 0};
  __builtin_amdgcn_tensor_load_to_lds(g0, g1, z4, z4, z8, 0);
}

// ---------------------------------------------------------------------------
// Stage 1: per-batch GEMM  out = W[M,K] * X[K,4096] + bias
// layout 0: pixel-major  out[(b*4096 + n)*M + m]        (pq, pk : M=64)
// layout 1: channel-major out[(b*M + m)*4096 + n]       (pv : M=512)
// out2 (optional): w-major pv_wm[b][w][c][g=h], g contiguous.
// block = 128 (4 waves), tile 64x64, K step 64, X transposed into LDS.
// Staging is batch-unrolled: 16 global_load_b128 (8 A + 8 B) issue back-to-back, then
// convert+store (f32 -> bf16 must go through VALU, so no TDM here).
// ---------------------------------------------------------------------------
__global__ void proj_gemm(const float* __restrict__ X, const float* __restrict__ Wm,
                          const float* __restrict__ bias, bf16_t* __restrict__ out,
                          bf16_t* __restrict__ out2, int M, int K, int layout) {
  const int N = 4096;
  const int n0 = blockIdx.x * 64;
  const int m0 = blockIdx.y * 64;
  const int b  = blockIdx.z;
  const int t = threadIdx.x;
  const int lane = t & 31, wv = t >> 5;
  const int am = lane & 15, half = lane >> 4;
  const int ms = wv * 16;

  __shared__ bf16_t As[64][LDP];    // [m][k] row-major
  __shared__ bf16_t BsT[64][LDP];   // [n][k] (transposed on store)

  // this thread's fixed staging coordinates: 8 rows apart, fixed quad column
  const int arow = t >> 4, aq = t & 15;              // A: rows arow+8j, col aq
  const float* Xb = X + (size_t)b * K * N;
  const float* wp = Wm + (size_t)(m0 + arow) * K + aq * 4;
  const float* xp = Xb + (size_t)arow * N + n0 + aq * 4;

  v8f acc[4] = {};

  for (int k0 = 0; k0 < K; k0 += 64) {
    float4 va[8], vb[8];
#pragma unroll
    for (int j = 0; j < 8; ++j)       // batch global loads: A tile
      va[j] = *(const float4*)(wp + (size_t)(8 * j) * K + k0);
#pragma unroll
    for (int j = 0; j < 8; ++j)       // batch global loads: B tile
      vb[j] = *(const float4*)(xp + (size_t)(k0 + 8 * j) * N);
    if (k0 + 64 < K)                  // prefetch next X tile
      __builtin_prefetch(xp + (size_t)(k0 + 64 + 8 * (t & 7)) * N, 0, 3);
#pragma unroll
    for (int j = 0; j < 8; ++j)       // convert + store A (row-major)
      *(uint2*)&As[arow + 8 * j][aq * 4] =
          make_uint2(pk2(va[j].x, va[j].y), pk2(va[j].z, va[j].w));
#pragma unroll
    for (int j = 0; j < 8; ++j) {     // convert + transpose-store B
      int kk = arow + 8 * j, n = aq * 4;
      BsT[n + 0][kk] = f2bf(vb[j].x);
      BsT[n + 1][kk] = f2bf(vb[j].y);
      BsT[n + 2][kk] = f2bf(vb[j].z);
      BsT[n + 3][kk] = f2bf(vb[j].w);
    }
    __syncthreads();

#pragma unroll
    for (int kk = 0; kk < 64; kk += 32) {
      v16bf a = afrag(&As[ms + am][0], kk, half);
#pragma unroll
      for (int nt = 0; nt < 4; ++nt) {
        v16bf bb = bfrag(&BsT[nt * 16 + am][0], kk, half);
        acc[nt] = wmma_bf16(a, bb, acc[nt]);
      }
    }
    __syncthreads();
  }

#pragma unroll
  for (int nt = 0; nt < 4; ++nt) {
#pragma unroll
    for (int v = 0; v < 8; ++v) {
      int m = m0 + ms + v + 8 * half;
      int n = n0 + nt * 16 + am;
      float r = acc[nt][v] + bias[m];
      bf16_t rb = f2bf(r);
      if (layout == 0)
        out[((size_t)b * N + n) * (size_t)M + m] = rb;
      else
        out[((size_t)b * M + m) * (size_t)N + n] = rb;
      if (out2)  // pv_wm[b][w][c][h]; n = h*64+w within the batch plane
        out2[(((size_t)b * 64 + (n & 63)) * 512 + m) * 64 + (n >> 6)] = rb;
    }
  }
}

// ---------------------------------------------------------------------------
// Stage 2: attention scores, 64x64x64 GEMM per block.
// mode 0 (eH): fixed (b, w=o), rows r = h;  E[h,g] -> e[b][h][o][g]
// mode 1 (eW): fixed (b, h=o), rows r = w;  E[w,f] -> e[b][o][w][f]
// Tiles fetched by the Tensor Data Mover straight into padded LDS.
// ---------------------------------------------------------------------------
__global__ void attn_scores(const bf16_t* __restrict__ pqT,
                            const bf16_t* __restrict__ pkT,
                            float* __restrict__ e, int mode) {
  const int o = blockIdx.x, b = blockIdx.y;
  const int t = threadIdx.x, lane = t & 31, wv = t >> 5;
  const int am = lane & 15, half = lane >> 4;
  const int ms = wv * 16;

  __shared__ bf16_t Aq[64][LDP];   // [m][k]
  __shared__ bf16_t Bk[64][LDP];   // [n][k]

  if (wv == 0) {
    size_t base = (mode == 0) ? ((size_t)(b * 64) * 64 + o) * 64
                              : ((size_t)(b * 64 + o) * 64) * 64;
    unsigned stride = (mode == 0) ? 4096u : 64u;   // elems between rows
    tdm_load_2d(pqT + base, &Aq[0][0], 64, 64, stride);
    tdm_load_2d(pkT + base, &Bk[0][0], 64, 64, stride);
    __builtin_amdgcn_s_wait_tensorcnt(0);
  }
  __syncthreads();

  v8f acc[4] = {};
#pragma unroll
  for (int k0 = 0; k0 < 64; k0 += 32) {
    v16bf a = afrag(&Aq[ms + am][0], k0, half);
#pragma unroll
    for (int nt = 0; nt < 4; ++nt) {
      v16bf bb = bfrag(&Bk[nt * 16 + am][0], k0, half);
      acc[nt] = wmma_bf16(a, bb, acc[nt]);
    }
  }

#pragma unroll
  for (int nt = 0; nt < 4; ++nt) {
#pragma unroll
    for (int v = 0; v < 8; ++v) {
      int m = ms + v + 8 * half;
      int n = nt * 16 + am;
      size_t off = (mode == 0)
          ? ((size_t)((b * 64 + m) * 64 + o)) * 64 + n
          : ((size_t)((b * 64 + o) * 64 + m)) * 64 + n;
      e[off] = acc[nt][v];
    }
  }
}

// ---------------------------------------------------------------------------
// Stage 3: joint softmax over concat(eH_row[64], eW_row[64]) per (b,h,w).
// One wave32 per row, 4 values per lane; diagonal (g==h) masked here.
// ---------------------------------------------------------------------------
__global__ void softmax_cc(const float* __restrict__ eH, const float* __restrict__ eW,
                           bf16_t* __restrict__ aH, bf16_t* __restrict__ aW) {
  const int wv = threadIdx.x >> 5, lane = threadIdx.x & 31;
  const int row = blockIdx.x * 8 + wv;   // (b*64 + h)*64 + w
  const int h = (row >> 6) & 63;

  const float* ph = eH + (size_t)row * 64;
  const float* pw = eW + (size_t)row * 64;
  float e0 = ph[2 * lane], e1 = ph[2 * lane + 1];
  if (2 * lane == h)     e0 = NINF;
  if (2 * lane + 1 == h) e1 = NINF;
  float f0 = pw[2 * lane], f1 = pw[2 * lane + 1];

  float mx = fmaxf(fmaxf(e0, e1), fmaxf(f0, f1));
#pragma unroll
  for (int s = 16; s > 0; s >>= 1) mx = fmaxf(mx, __shfl_xor(mx, s, 32));

  float s0 = __expf(e0 - mx), s1 = __expf(e1 - mx);
  float s2 = __expf(f0 - mx), s3 = __expf(f1 - mx);
  float sum = s0 + s1 + s2 + s3;
#pragma unroll
  for (int s = 16; s > 0; s >>= 1) sum += __shfl_xor(sum, s, 32);
  float inv = 1.0f / sum;

  aH[(size_t)row * 64 + 2 * lane]     = f2bf(s0 * inv);
  aH[(size_t)row * 64 + 2 * lane + 1] = f2bf(s1 * inv);
  aW[(size_t)row * 64 + 2 * lane]     = f2bf(s2 * inv);
  aW[(size_t)row * 64 + 2 * lane + 1] = f2bf(s3 * inv);
}

// ---------------------------------------------------------------------------
// Stage 4a: OutH, fixed (b,w): Out[c,h] = sum_g pv[b,c,g,w]*attH[b,h,w,g]
// A from global pv_wm[b][w][c][g] (g contiguous), software-pipelined across
// the 4 c-subtiles; B tile TDM-loaded into LDS [n=h][k=g].
// Writes out = gamma*OutH + value.
// ---------------------------------------------------------------------------
__global__ void av_h(const bf16_t* __restrict__ pv_wm, const bf16_t* __restrict__ aH,
                     const float* __restrict__ value, const float* __restrict__ gamma,
                     float* __restrict__ out) {
  const int w = blockIdx.x, b = blockIdx.y;
  const int t = threadIdx.x, lane = t & 31, wv = t >> 5;
  const int am = lane & 15, half = lane >> 4;

  __shared__ bf16_t att[64][LDP];   // [h][g]
  if (wv == 0) {
    tdm_load_2d(aH + ((size_t)(b * 64) * 64 + w) * 64, &att[0][0], 64, 64, 4096u);
    __builtin_amdgcn_s_wait_tensorcnt(0);
  }
  __syncthreads();
  const float g = gamma[0];

  const bf16_t* base = pv_wm + (((size_t)b * 64 + w) * 512 + wv * 64 + am) * 64;
  v16bf a0 = afrag(base, 0, half), a1 = afrag(base, 32, half);
  for (int s = 0; s < 4; ++s) {
    v16bf na0 = a0, na1 = a1;
    if (s < 3) {  // prefetch next subtile's A fragments during compute
      const bf16_t* np = base + (size_t)(s + 1) * 16 * 64;
      na0 = afrag(np, 0, half);
      na1 = afrag(np, 32, half);
    }
    v8f acc[4] = {};
#pragma unroll
    for (int nt = 0; nt < 4; ++nt) {
      acc[nt] = wmma_bf16(a0, bfrag(&att[nt * 16 + am][0], 0, half), acc[nt]);
      acc[nt] = wmma_bf16(a1, bfrag(&att[nt * 16 + am][0], 32, half), acc[nt]);
    }
    const int c0 = wv * 64 + s * 16;
#pragma unroll
    for (int nt = 0; nt < 4; ++nt) {
#pragma unroll
      for (int v = 0; v < 8; ++v) {
        int c = c0 + v + 8 * half;
        int n = nt * 16 + am;                  // n = h
        size_t off = ((size_t)(b * 512 + c)) * 4096 + (size_t)n * 64 + w;
        out[off] = g * acc[nt][v] + value[off];
      }
    }
    a0 = na0; a1 = na1;
  }
}

// ---------------------------------------------------------------------------
// Stage 4b: OutW, fixed (b,h): Out[c,w] = sum_f pv[b,c,h,f]*attW[b,h,w,f]
// A from global pv channel-major rows (f contiguous), software-pipelined;
// B tile TDM-loaded into LDS [n=w][k=f]. Accumulates out += gamma*OutW.
// ---------------------------------------------------------------------------
__global__ void av_w(const bf16_t* __restrict__ pv, const bf16_t* __restrict__ aW,
                     const float* __restrict__ gamma, float* __restrict__ out) {
  const int h = blockIdx.x, b = blockIdx.y;
  const int t = threadIdx.x, lane = t & 31, wv = t >> 5;
  const int am = lane & 15, half = lane >> 4;

  __shared__ bf16_t att[64][LDP];   // [w][f]
  if (wv == 0) {
    tdm_load_2d(aW + ((size_t)(b * 64 + h) * 64) * 64, &att[0][0], 64, 64, 64u);
    __builtin_amdgcn_s_wait_tensorcnt(0);
  }
  __syncthreads();
  const float g = gamma[0];

  const bf16_t* base = pv + ((size_t)(b * 512 + wv * 64 + am)) * 4096 + (size_t)h * 64;
  v16bf a0 = afrag(base, 0, half), a1 = afrag(base, 32, half);
  for (int s = 0; s < 4; ++s) {
    v16bf na0 = a0, na1 = a1;
    if (s < 3) {  // prefetch next subtile's A fragments during compute
      const bf16_t* np = base + (size_t)(s + 1) * 16 * 4096;
      na0 = afrag(np, 0, half);
      na1 = afrag(np, 32, half);
    }
    v8f acc[4] = {};
#pragma unroll
    for (int nt = 0; nt < 4; ++nt) {
      acc[nt] = wmma_bf16(a0, bfrag(&att[nt * 16 + am][0], 0, half), acc[nt]);
      acc[nt] = wmma_bf16(a1, bfrag(&att[nt * 16 + am][0], 32, half), acc[nt]);
    }
    const int c0 = wv * 64 + s * 16;
#pragma unroll
    for (int nt = 0; nt < 4; ++nt) {
#pragma unroll
      for (int v = 0; v < 8; ++v) {
        int c = c0 + v + 8 * half;
        int n = nt * 16 + am;                  // n = w
        size_t off = ((size_t)(b * 512 + c)) * 4096 + (size_t)h * 64 + n;
        out[off] = out[off] + g * acc[nt][v];
      }
    }
    a0 = na0; a1 = na1;
  }
}

// ---------------------------------------------------------------------------
extern "C" void kernel_launch(void* const* d_in, const int* in_sizes, int n_in,
                              void* d_out, int out_size, void* d_ws, size_t ws_size,
                              hipStream_t stream) {
  (void)in_sizes; (void)n_in; (void)out_size; (void)ws_size;
  const float* query = (const float*)d_in[0];
  const float* key_t = (const float*)d_in[1];
  const float* value = (const float*)d_in[2];
  const float* Wq    = (const float*)d_in[3];
  const float* bq    = (const float*)d_in[4];
  const float* Wk    = (const float*)d_in[5];
  const float* bk    = (const float*)d_in[6];
  const float* Wv    = (const float*)d_in[7];
  const float* bv    = (const float*)d_in[8];
  const float* gamma = (const float*)d_in[9];
  float* out = (float*)d_out;

  char* ws = (char*)d_ws;                        // 96 MB total
  bf16_t* pqT  = (bf16_t*)(ws);                  //  4 MB  [b][h][w][64]
  bf16_t* pkT  = (bf16_t*)(ws + (4ull  << 20));  //  4 MB
  bf16_t* pv   = (bf16_t*)(ws + (8ull  << 20));  // 32 MB  [b][512][4096]
  bf16_t* pvwm = (bf16_t*)(ws + (40ull << 20));  // 32 MB  [b][w][c][h]
  float*  eH   = (float*) (ws + (72ull << 20));  //  8 MB  [b][h][w][g]
  float*  eW   = (float*) (ws + (80ull << 20));  //  8 MB  [b][h][w][f]
  bf16_t* aH   = (bf16_t*)(ws + (88ull << 20));  //  4 MB
  bf16_t* aW   = (bf16_t*)(ws + (92ull << 20));  //  4 MB

  // Stage 1: projections (bf16 WMMA GEMMs)
  proj_gemm<<<dim3(64, 1, 8), 128, 0, stream>>>(query, Wq, bq, pqT, nullptr, 64, 512, 0);
  proj_gemm<<<dim3(64, 1, 8), 128, 0, stream>>>(key_t, Wk, bk, pkT, nullptr, 64, 512, 0);
  proj_gemm<<<dim3(64, 8, 8), 128, 0, stream>>>(value, Wv, bv, pv, pvwm, 512, 512, 1);

  // Stage 2: criss-cross scores (TDM-staged tiles)
  attn_scores<<<dim3(64, 8), 128, 0, stream>>>(pqT, pkT, eH, 0);
  attn_scores<<<dim3(64, 8), 128, 0, stream>>>(pqT, pkT, eW, 1);

  // Stage 3: joint softmax with diagonal mask
  softmax_cc<<<dim3(4096), 256, 0, stream>>>(eH, eW, aH, aW);

  // Stage 4: attention-weighted value aggregation + residual
  av_h<<<dim3(64, 8), 256, 0, stream>>>(pvwm, aH, value, gamma, out);
  av_w<<<dim3(64, 8), 256, 0, stream>>>(pv, aW, gamma, out);
}
